// FeatureAttention_47141561041580
// MI455X (gfx1250) — matrix operand order repaired and checked by reading, not compiled, not run on hardware
//
#include <hip/hip_runtime.h>
#include <hip/hip_bf16.h>

// ---------------------------------------------------------------------------
// Dual (feature/time) attention + residual layernorm for B=32, N=T=F=1024.
// bf16 WMMA GEMMs (v_wmma_f32_16x16x32_bf16) with fp32 accumulate; Tensor
// Data Mover (tensor_load_to_lds) feeds LDS tiles with hardware padding.
// All GEMMs are C = A * B^T with B^T row-major -> coalesced/linear tiles.
// ---------------------------------------------------------------------------

typedef __attribute__((ext_vector_type(16))) __bf16 v16bf;
typedef __attribute__((ext_vector_type(8)))  float  v8f;
typedef __attribute__((ext_vector_type(4)))  unsigned int v4u;
typedef __attribute__((ext_vector_type(8)))  int    v8i;
typedef __attribute__((ext_vector_type(4)))  int    v4i;

#define N_DIM 1024
#define LDP   40       // padded LDS row length (elements), 80B pitch

#if __has_builtin(__builtin_amdgcn_tensor_load_to_lds) && \
    __has_builtin(__builtin_amdgcn_s_wait_tensorcnt)
#define USE_TDM 1
#else
#define USE_TDM 0
#endif

__device__ __forceinline__ v16bf as_v16bf(uint4 lo, uint4 hi) {
    union { uint4 u[2]; v16bf v; } t;
    t.u[0] = lo; t.u[1] = hi;
    return t.v;
}

#if USE_TDM
// Issue a TDM load of a 128-row x 32-elem bf16 tile (row stride 1024 elems)
// into LDS at lds_off, padding each 64B row to an 80B pitch (LDP elements).
__device__ __forceinline__ void tdm_load_tile(const __bf16* gsrc, unsigned lds_off) {
    const unsigned long long ga = (unsigned long long)(uintptr_t)gsrc;
    v4u g0;
    g0[0] = 1u;                                   // count=1, user-mode flags 0
    g0[1] = lds_off;                              // lds_addr (bytes)
    g0[2] = (unsigned)(ga & 0xFFFFFFFFu);         // global_addr[31:0]
    g0[3] = (unsigned)((ga >> 32) & 0x01FFFFFFu)  // global_addr[56:32]
          | (2u << 30);                           // type = 2 ("image")
    v8i g1;
    g1[0] = (int)((1u << 16)                      // data_size = 2 bytes
                | (1u << 20)                      // pad_enable
                | (3u << 22)                      // pad_interval: 16 DWORDs (64B)
                | (3u << 25));                    // pad_amount:   4 DWORDs (16B)
    g1[1] = (int)(0x0400u << 16);                 // tensor_dim0 = 1024 (lo16)
    g1[2] = (int)(0x0400u << 16);                 // dim0 hi=0 | tensor_dim1=1024 (lo16)
    g1[3] = (int)(32u << 16);                     // dim1 hi=0 | tile_dim0 = 32
    g1[4] = (int)128;                             // tile_dim1 = 128 | tile_dim2 = 0
    g1[5] = (int)1024;                            // tensor_dim0_stride = 1024
    g1[6] = 0;
    g1[7] = 0;
    const v4i z4 = {0, 0, 0, 0};                  // groups 2/3 unused (2-D tile)
    const v8i z8 = {0, 0, 0, 0, 0, 0, 0, 0};
    __builtin_amdgcn_tensor_load_to_lds(g0, g1, z4, z4, z8, 0);
}
#endif

// ---------------------------------------------------------------------------
// Batched GEMM: C[b] = A[b] (1024x1024 bf16 row-major)
//                     * BT[b]^T (BT row-major bf16, batch stride strideBT)
// Block tile 128x128, K-step 32, 8 waves (2x4), double-buffered LDS.
// ---------------------------------------------------------------------------
template <bool OUT_BF16>
__global__ __launch_bounds__(256)
void gemm_abt_kernel(const __bf16* __restrict__ A,
                     const __bf16* __restrict__ BT,
                     void* __restrict__ Cv,
                     unsigned long long strideBT) {
    __shared__ __bf16 sA[2][128 * LDP];
    __shared__ __bf16 sB[2][128 * LDP];

    const int tid  = threadIdx.x;
    const int lane = tid & 31;
    const int wave = tid >> 5;
    const int wm   = wave >> 2;       // 0..1 -> 64 rows each
    const int wn   = wave & 3;        // 0..3 -> 32 cols each
    const int half = lane >> 4;       // K-half selector per WMMA layout

    const int m0 = blockIdx.x * 128;
    const int n0 = blockIdx.y * 128;
    const size_t bz = (size_t)blockIdx.z;

    const __bf16* gAbase = A + (bz << 20) + (size_t)m0 * N_DIM;
    const __bf16* gBbase = BT + bz * strideBT + (size_t)n0 * N_DIM;

    v8f acc[4][2] = {};
    int buf = 0;

#if USE_TDM
    const unsigned ldsA[2] = { (unsigned)(uintptr_t)&sA[0][0],
                               (unsigned)(uintptr_t)&sA[1][0] };
    const unsigned ldsB[2] = { (unsigned)(uintptr_t)&sB[0][0],
                               (unsigned)(uintptr_t)&sB[1][0] };
    if (wave == 0) {                          // one DMA per workgroup tile
        tdm_load_tile(gAbase, ldsA[0]);
        tdm_load_tile(gBbase, ldsB[0]);
    }
#else
    const uint4* gA = (const uint4*)gAbase;
    const uint4* gB = (const uint4*)gBbase;
    const int r0 = tid >> 2, r1 = r0 + 64, c0 = tid & 3;
    uint4 ra0 = gA[r0 * 128 + c0];
    uint4 ra1 = gA[r1 * 128 + c0];
    uint4 rb0 = gB[r0 * 128 + c0];
    uint4 rb1 = gB[r1 * 128 + c0];
#endif

    #pragma unroll 1
    for (int kt = 0; kt < 32; ++kt) {
#if USE_TDM
        if (wave == 0) __builtin_amdgcn_s_wait_tensorcnt(0);  // tile kt landed
        __syncthreads();
        if (wave == 0 && kt + 1 < 32) {       // overlap DMA of next tile
            tdm_load_tile(gAbase + (kt + 1) * 32, ldsA[buf ^ 1]);
            tdm_load_tile(gBbase + (kt + 1) * 32, ldsB[buf ^ 1]);
        }
#else
        *(uint4*)&sA[buf][r0 * LDP + c0 * 8] = ra0;
        *(uint4*)&sA[buf][r1 * LDP + c0 * 8] = ra1;
        *(uint4*)&sB[buf][r0 * LDP + c0 * 8] = rb0;
        *(uint4*)&sB[buf][r1 * LDP + c0 * 8] = rb1;
        __syncthreads();
        if (kt + 1 < 32) {
            const int ko = (kt + 1) * 4 + c0;
            ra0 = gA[r0 * 128 + ko];
            ra1 = gA[r1 * 128 + ko];
            rb0 = gB[r0 * 128 + ko];
            rb1 = gB[r1 * 128 + ko];
        }
        if (kt + 2 < 32) {
            const int ko2 = (kt + 2) * 4 + c0;
            __builtin_prefetch(&gA[r0 * 128 + ko2], 0, 3);
            __builtin_prefetch(&gB[r1 * 128 + ko2], 0, 3);
        }
#endif

        // B fragments: 32x16 bf16; lane<16: K0-7 & 16-23; lane>=16: K8-15 & 24-31, N=lane%16
        v16bf bfr[2];
        #pragma unroll
        for (int j = 0; j < 2; ++j) {
            const int n = wn * 32 + j * 16 + (lane & 15);
            const uint4* p = (const uint4*)&sB[buf][n * LDP + half * 8];
            bfr[j] = as_v16bf(p[0], p[2]);
        }

        // A fragments: 16x32 bf16; lane<16: K0-7 & 16-23; lane>=16: K8-15 & 24-31
        #pragma unroll
        for (int i = 0; i < 4; ++i) {
            const int m = wm * 64 + i * 16 + (lane & 15);
            const uint4 lo = *(const uint4*)&sA[buf][m * LDP + half * 8];
            const uint4 hi = *(const uint4*)&sA[buf][m * LDP + 16 + half * 8];
            const v16bf afr = as_v16bf(lo, hi);
            acc[i][0] = __builtin_amdgcn_wmma_f32_16x16x32_bf16(
                false, afr, false, bfr[0], (short)0, acc[i][0], false, false);
            acc[i][1] = __builtin_amdgcn_wmma_f32_16x16x32_bf16(
                false, afr, false, bfr[1], (short)0, acc[i][1], false, false);
        }

        buf ^= 1;
        __syncthreads();
    }

    // epilogue: VGPR r holds M=r (lanes 0-15) / M=r+8 (lanes 16-31); N=lane%16
    const size_t cbase = bz << 20;
    #pragma unroll
    for (int i = 0; i < 4; ++i) {
        const int mrow = m0 + wm * 64 + i * 16 + half * 8;
        #pragma unroll
        for (int j = 0; j < 2; ++j) {
            const int ncol = n0 + wn * 32 + j * 16 + (lane & 15);
            #pragma unroll
            for (int r = 0; r < 8; ++r) {
                const size_t idx = cbase + (size_t)(mrow + r) * N_DIM + ncol;
                if (OUT_BF16) ((__bf16*)Cv)[idx] = (__bf16)acc[i][j][r];
                else          ((float*)Cv)[idx]  = acc[i][j][r];
            }
        }
    }
}

// ---------------------------------------------------------------------------
// fp32 -> bf16 pack: straight copy (optional) + transposed copy via LDS tile.
// ---------------------------------------------------------------------------
__global__ __launch_bounds__(256)
void pack_kernel(const float* __restrict__ in,
                 __bf16* __restrict__ outS,
                 __bf16* __restrict__ outT,
                 int batched) {
    __shared__ float tile[32][33];
    const int bx = blockIdx.x * 32;
    const int by = blockIdx.y * 32;
    const size_t boff = batched ? ((size_t)blockIdx.z << 20) : 0;
    const int tx = threadIdx.x & 31;
    const int ty = threadIdx.x >> 5;

    #pragma unroll
    for (int k = 0; k < 4; ++k) {
        const int r = ty + k * 8;
        const float v = in[boff + (size_t)(by + r) * N_DIM + bx + tx];
        tile[r][tx] = v;
        if (outS) outS[boff + (size_t)(by + r) * N_DIM + bx + tx] = (__bf16)v;
    }
    __syncthreads();
    #pragma unroll
    for (int k = 0; k < 4; ++k) {
        const int r = ty + k * 8;
        outT[boff + (size_t)(bx + r) * N_DIM + by + tx] = (__bf16)tile[tx][r];
    }
}

// ---------------------------------------------------------------------------
// block-wide reductions (8 wave32's)
// ---------------------------------------------------------------------------
__device__ __forceinline__ float block_sum(float v, float* red) {
    #pragma unroll
    for (int off = 16; off; off >>= 1) v += __shfl_xor(v, off, 32);
    __syncthreads();
    if ((threadIdx.x & 31) == 0) red[threadIdx.x >> 5] = v;
    __syncthreads();
    float t = red[0];
    #pragma unroll
    for (int k = 1; k < 8; ++k) t += red[k];
    return t;
}

__device__ __forceinline__ float block_max(float v, float* red) {
    #pragma unroll
    for (int off = 16; off; off >>= 1) v = fmaxf(v, __shfl_xor(v, off, 32));
    __syncthreads();
    if ((threadIdx.x & 31) == 0) red[threadIdx.x >> 5] = v;
    __syncthreads();
    float t = red[0];
    #pragma unroll
    for (int k = 1; k < 8; ++k) t = fmaxf(t, red[k]);
    return t;
}

// Row softmax: fp32 scores in, bf16 probabilities out. One block per row.
__global__ __launch_bounds__(256)
void softmax_kernel(const float* __restrict__ S, __bf16* __restrict__ P) {
    __shared__ float red[8];
    const size_t row = (size_t)blockIdx.x << 10;
    const int tid = threadIdx.x;
    float v[4];
    float m = -1e30f;
    #pragma unroll
    for (int i = 0; i < 4; ++i) {
        v[i] = S[row + tid + (i << 8)];
        m = fmaxf(m, v[i]);
    }
    m = block_max(m, red);
    float s = 0.f;
    #pragma unroll
    for (int i = 0; i < 4; ++i) { v[i] = __expf(v[i] - m); s += v[i]; }
    s = block_sum(s, red);
    const float inv = 1.0f / s;
    #pragma unroll
    for (int i = 0; i < 4; ++i)
        P[row + tid + (i << 8)] = (__bf16)(v[i] * inv);
}

// combined = 0.5*(ctxA + ctxB) + x ; out = LN(combined)*gamma + beta
__global__ __launch_bounds__(256)
void ln_kernel(const float* __restrict__ x, const float* __restrict__ cA,
               const float* __restrict__ cB, const float* __restrict__ gamma,
               const float* __restrict__ beta, float* __restrict__ out) {
    __shared__ float red[8];
    const size_t row = (size_t)blockIdx.x << 10;
    const int tid = threadIdx.x;
    float h[4];
    float s = 0.f;
    #pragma unroll
    for (int i = 0; i < 4; ++i) {
        const size_t idx = row + tid + (i << 8);
        h[i] = 0.5f * (cA[idx] + cB[idx]) + x[idx];
        s += h[i];
    }
    const float mean = block_sum(s, red) * (1.0f / 1024.0f);
    float s2 = 0.f;
    #pragma unroll
    for (int i = 0; i < 4; ++i) { const float d = h[i] - mean; s2 += d * d; }
    const float var = block_sum(s2, red) * (1.0f / 1024.0f);
    const float rs = rsqrtf(var + 1e-6f);
    #pragma unroll
    for (int i = 0; i < 4; ++i) {
        const int f = tid + (i << 8);
        out[row + f] = (h[i] - mean) * rs * gamma[f] + beta[f];
    }
}

// ---------------------------------------------------------------------------
extern "C" void kernel_launch(void* const* d_in, const int* in_sizes, int n_in,
                              void* d_out, int out_size, void* d_ws, size_t ws_size,
                              hipStream_t stream) {
    (void)in_sizes; (void)n_in; (void)out_size; (void)ws_size;
    const float* x     = (const float*)d_in[0];
    const float* Wf    = (const float*)d_in[1];
    const float* Wt    = (const float*)d_in[2];
    const float* gamma = (const float*)d_in[3];
    const float* beta  = (const float*)d_in[4];
    float* out = (float*)d_out;

    const size_t MB = 1ull << 20;
    const unsigned long long NN = 1024ull * 1024ull;
    char* ws = (char*)d_ws;
    __bf16* x_bf   = (__bf16*)(ws);               //  64 MB
    __bf16* xt_bf  = (__bf16*)(ws + 64 * MB);     //  64 MB
    __bf16* WfT    = (__bf16*)(ws + 128 * MB);    //   2 MB
    __bf16* WtT    = (__bf16*)(ws + 130 * MB);    //   2 MB
    __bf16* q_bf   = (__bf16*)(ws + 132 * MB);    //  64 MB
    __bf16* attn   = (__bf16*)(ws + 196 * MB);    //  64 MB
    float*  scores = (float*)(ws + 260 * MB);     // 128 MB
    float*  ctxA   = (float*)(ws + 388 * MB);     // 128 MB (time_ctx)
    float*  ctxB   = (float*)(ws + 516 * MB);     // 128 MB (feat_ctx)

    pack_kernel<<<dim3(32, 32, 32), 256, 0, stream>>>(x, x_bf, xt_bf, 1);
    pack_kernel<<<dim3(32, 32, 1), 256, 0, stream>>>(Wf, nullptr, WfT, 0);
    pack_kernel<<<dim3(32, 32, 1), 256, 0, stream>>>(Wt, nullptr, WtT, 0);

    const dim3 gg(8, 8, 32);

    // ---- time branch: qt = x@Wt ; At = softmax(qt@x^T) ; ctxA = At@x ----
    gemm_abt_kernel<true ><<<gg, 256, 0, stream>>>(x_bf, WtT, q_bf, 0ull);
    gemm_abt_kernel<false><<<gg, 256, 0, stream>>>(q_bf, x_bf, scores, NN);
    softmax_kernel<<<32 * 1024, 256, 0, stream>>>(scores, attn);
    gemm_abt_kernel<false><<<gg, 256, 0, stream>>>(attn, xt_bf, ctxA, NN);

    // ---- feature branch: qf = x^T@Wf ; Af = softmax(qf@x) ; ctxB = x@Af^T ----
    gemm_abt_kernel<true ><<<gg, 256, 0, stream>>>(xt_bf, WfT, q_bf, 0ull);
    gemm_abt_kernel<false><<<gg, 256, 0, stream>>>(q_bf, xt_bf, scores, NN);
    softmax_kernel<<<32 * 1024, 256, 0, stream>>>(scores, attn);
    gemm_abt_kernel<false><<<gg, 256, 0, stream>>>(x_bf, attn, ctxB, NN);

    // ---- residual + layernorm ----
    ln_kernel<<<32 * 1024, 256, 0, stream>>>(x, ctxA, ctxB, gamma, beta, out);
}
